// GAttn_28398323761582
// MI455X (gfx1250) — compile-verified, matrix operand order checked by reading, not run on hardware
//
#include <hip/hip_runtime.h>
#include <hip/hip_bf16.h>
#include <stdint.h>

typedef __attribute__((ext_vector_type(16))) _Float16 v16h;
typedef __attribute__((ext_vector_type(8)))  float    v8f;

#define BN_EPS 1e-5f

// Issue one wave-wide async global->LDS 16B-per-lane copy (512B per wave op).
__device__ __forceinline__ void async_copy_b128(const void* gptr, void* lds_ptr)
{
    unsigned ldsoff = (unsigned)(uintptr_t)lds_ptr;
    asm volatile("global_load_async_to_lds_b128 %0, %1, off"
                 :: "v"(ldsoff), "v"(gptr) : "memory");
}
__device__ __forceinline__ void async_wait_all()
{
    asm volatile("s_wait_asynccnt 0x0" ::: "memory");
}

// ---------------------------------------------------------------------------
// Kernel A: fold BatchNorm into weights; produce f16 weight copies + f32 bias
// ---------------------------------------------------------------------------
__global__ void fold_weights_kernel(
    const float* __restrict__ w_qk, const float* __restrict__ qk_g, const float* __restrict__ qk_b,
    const float* __restrict__ qk_m, const float* __restrict__ qk_v,
    const float* __restrict__ w_v,  const float* __restrict__ v_g,  const float* __restrict__ v_b,
    const float* __restrict__ v_m,  const float* __restrict__ v_v,
    const float* __restrict__ w_pe, const float* __restrict__ pe_g, const float* __restrict__ pe_b,
    const float* __restrict__ pe_m, const float* __restrict__ pe_v,
    const float* __restrict__ w_pr, const float* __restrict__ pr_g, const float* __restrict__ pr_b,
    const float* __restrict__ pr_m, const float* __restrict__ pr_v,
    _Float16* __restrict__ wcat16, float* __restrict__ bias_cat,
    _Float16* __restrict__ wproj16, float* __restrict__ bias_pr,
    float* __restrict__ wpe_s, float* __restrict__ bias_pe)
{
    const int idx = blockIdx.x * blockDim.x + threadIdx.x;
    const int stride = gridDim.x * blockDim.x;

    for (int i = idx; i < 768 * 256; i += stride) {
        int o = i >> 8, c = i & 255;
        float s, w;
        if (o < 512) { s = qk_g[o] * rsqrtf(qk_v[o] + BN_EPS); w = w_qk[o * 256 + c]; }
        else { int oo = o - 512; s = v_g[oo] * rsqrtf(v_v[oo] + BN_EPS); w = w_v[oo * 256 + c]; }
        wcat16[i] = (_Float16)(w * s);
    }
    for (int o = idx; o < 768; o += stride) {
        if (o < 512) { float s = qk_g[o] * rsqrtf(qk_v[o] + BN_EPS); bias_cat[o] = qk_b[o] - qk_m[o] * s; }
        else { int oo = o - 512; float s = v_g[oo] * rsqrtf(v_v[oo] + BN_EPS); bias_cat[o] = v_b[oo] - v_m[oo] * s; }
    }
    for (int i = idx; i < 256 * 256; i += stride) {
        int o = i >> 8;
        float s = pr_g[o] * rsqrtf(pr_v[o] + BN_EPS);
        wproj16[i] = (_Float16)(w_pr[i] * s);
    }
    for (int o = idx; o < 256; o += stride) {
        float s = pr_g[o] * rsqrtf(pr_v[o] + BN_EPS);
        bias_pr[o] = pr_b[o] - pr_m[o] * s;
        float spe = pe_g[o] * rsqrtf(pe_v[o] + BN_EPS);
        bias_pe[o] = pe_b[o] - pe_m[o] * spe;
    }
    for (int i = idx; i < 256 * 25; i += stride) {
        int c = i / 25;
        float s = pe_g[c] * rsqrtf(pe_v[c] + BN_EPS);
        wpe_s[i] = w_pe[i] * s;
    }
}

// ---------------------------------------------------------------------------
// Kernel B: pack x (B,C,H,W) f32 -> X16 row-major (B*N, C) f16
// ---------------------------------------------------------------------------
__global__ void pack_x_kernel(const float* __restrict__ x, _Float16* __restrict__ X16)
{
    int i = blockIdx.x * blockDim.x + threadIdx.x;
    if (i >= 2 * 256 * 4096) return;
    int c = i & 255;
    int n = (i >> 8) & 4095;
    int b = i >> 20;
    X16[i] = (_Float16)x[((size_t)(b * 256 + c) << 12) + n];
}

// ---------------------------------------------------------------------------
// Kernel C: fused QKV GEMM: X16 (8192x256) * wcat16^T + bias.
// 128-column weight slab (64 KB f16) async-staged once per block into LDS;
// K-loop streams only A from global, B from ds_load.
// ---------------------------------------------------------------------------
__global__ __launch_bounds__(256) void qkv_gemm_kernel(
    const _Float16* __restrict__ X16, const _Float16* __restrict__ W,
    const float* __restrict__ bias,
    _Float16* __restrict__ q16, _Float16* __restrict__ k16,
    _Float16* __restrict__ v16t, float* __restrict__ vf32)
{
    __shared__ _Float16 wlds[128 * 256];       // 64 KB weight slab
    const int lane = threadIdx.x & 31;
    const int wave = threadIdx.x >> 5;
    const int wm = wave & 3, wn = wave >> 2;
    const int m0 = blockIdx.x * 64 + wm * 16;
    const int n0 = wn * 64;                    // within-slab column base
    const int nslab = blockIdx.y * 128;        // global column base of slab
    const int row = lane & 15, hi = lane >> 4;

    // async stage the weight slab: 256 threads x 16 x 16B = 64 KB
    {
        const _Float16* wslab = W + (size_t)nslab * 256;
#pragma unroll
        for (int i = 0; i < 16; ++i) {
            int off = (threadIdx.x + i * 256) * 8;   // halves
            async_copy_b128(wslab + off, wlds + off);
        }
        async_wait_all();
        __syncthreads();
    }

    v8f acc[4] = {};
    const _Float16* arow = X16 + (size_t)(m0 + row) * 256 + hi * 16;
    for (int k0 = 0; k0 < 256; k0 += 32) {
        __builtin_prefetch(arow + k0 + 64, 0, 1);
        v16h a = *(const v16h*)(arow + k0);
#pragma unroll
        for (int t = 0; t < 4; ++t) {
            v16h bm = *(const v16h*)(wlds + (n0 + t * 16 + row) * 256 + k0 + hi * 16);
            acc[t] = __builtin_amdgcn_wmma_f32_16x16x32_f16(
                false, a, false, bm, (short)0, acc[t], false, false);
        }
    }

#pragma unroll
    for (int t = 0; t < 4; ++t) {
        int o = nslab + n0 + t * 16 + row;      // output channel
        float bo = bias[o];
#pragma unroll
        for (int r = 0; r < 8; ++r) {
            int m = m0 + r + 8 * hi;            // global spatial row (b*4096 + n)
            float val = acc[t][r] + bo;
            int b = m >> 12, n = m & 4095;
            int ba = (b << 2) + (n >> 10), na = n & 1023;
            if (o < 256) {                      // q
                int h = o >> 5, d = o & 31;
                q16[(((size_t)((ba << 3) + h) << 10) + na) * 32 + d] = (_Float16)val;
            } else if (o < 512) {               // k
                int oo = o - 256, h = oo >> 5, d = oo & 31;
                k16[(((size_t)((ba << 3) + h) << 10) + na) * 32 + d] = (_Float16)val;
            } else {                            // v : f32 NCHW + transposed f16
                int c = o - 512;
                vf32[((size_t)(b * 256 + c) << 12) + n] = val;
                int h = c >> 5, d = c & 31;
                v16t[((size_t)((((ba << 3) + h) << 5) + d) << 10) + na] = (_Float16)val;
            }
        }
    }
}

// ---------------------------------------------------------------------------
// Kernel D: depthwise 5x5 conv on v (pad 2), BN folded. pp layout [b][c][n].
// ---------------------------------------------------------------------------
__global__ void pe_conv_kernel(const float* __restrict__ vf32,
                               const float* __restrict__ wpe,
                               const float* __restrict__ bias_pe,
                               float* __restrict__ pp)
{
    int i = blockIdx.x * blockDim.x + threadIdx.x;
    if (i >= 2 * 256 * 4096) return;
    int n = i & 4095;
    int c = (i >> 12) & 255;
    int b = i >> 20;
    int y = n >> 6, xq = n & 63;
    const float* vin = vf32 + ((size_t)(b * 256 + c) << 12);
    const float* w = wpe + c * 25;
    float acc = bias_pe[c];
#pragma unroll
    for (int dy = -2; dy <= 2; ++dy) {
        int yy = y + dy;
        if ((unsigned)yy >= 64u) continue;
#pragma unroll
        for (int dx = -2; dx <= 2; ++dx) {
            int xx = xq + dx;
            if ((unsigned)xx >= 64u) continue;
            acc += vin[yy * 64 + xx] * w[(dy + 2) * 5 + (dx + 2)];
        }
    }
    pp[i] = acc;
}

// ---------------------------------------------------------------------------
// Kernel E: flash attention per (ba, h, n-tile). K/V chunks async-staged
// into LDS shared by all 8 waves; hd=32 => one WMMA per 16x16 score tile.
// Epilogue fuses (o + pp) -> comb16 rows (B*N, C) f16 for the proj GEMM.
// ---------------------------------------------------------------------------
__global__ __launch_bounds__(256) void attn_kernel(
    const _Float16* __restrict__ q16, const _Float16* __restrict__ k16,
    const _Float16* __restrict__ v16t, const float* __restrict__ pp,
    _Float16* __restrict__ comb16)
{
    __shared__ _Float16 ldsK[32 * 32];       // K chunk: [m_local][d]   (2 KB)
    __shared__ _Float16 ldsV[32 * 32];       // V chunk: [d][m_local]   (2 KB)
    __shared__ _Float16 pbuf[8][16 * 32];    // per-wave P staging      (8 KB)

    const int lane = threadIdx.x & 31, wave = threadIdx.x >> 5;
    const int ba = blockIdx.z, h = blockIdx.y;
    const int n0 = blockIdx.x * 128 + wave * 16;
    const int row = lane & 15, hi = lane >> 4;
    const float scale = 0.17677669529663687f;   // 1/sqrt(hd=32)

    const _Float16* qbase = q16 + ((size_t)(ba * 8 + h) << 10) * 32;
    const _Float16* kbase = k16 + ((size_t)(ba * 8 + h) << 10) * 32;
    const _Float16* vbase = v16t + ((size_t)(ba * 8 + h) << 5) * 1024;

    // A operand: 16 query rows x K=32 (entire head dim), loaded once.
    v16h aq = *(const v16h*)(qbase + (size_t)(n0 + row) * 32 + hi * 16);

    float rmax[8], rsum[8];
#pragma unroll
    for (int r = 0; r < 8; ++r) { rmax[r] = -1e30f; rsum[r] = 0.f; }
    v8f o0 = {}, o1 = {};
    _Float16* pl = &pbuf[wave][0];

    for (int m0 = 0; m0 < 1024; m0 += 32) {
        __syncthreads();   // staging + P buffers free for reuse
        // Async stage: waves 0-3 copy K chunk (contiguous 2 KB),
        //              waves 4-7 copy V chunk (32 d-rows x 64 B).
        if (threadIdx.x < 128) {
            int off = threadIdx.x * 8;                       // halves
            async_copy_b128(kbase + (size_t)m0 * 32 + off, ldsK + off);
        } else {
            int tt = threadIdx.x - 128;
            int d = tt >> 2, part = tt & 3;
            async_copy_b128(vbase + (size_t)d * 1024 + m0 + part * 8,
                            ldsV + d * 32 + part * 8);
        }
        async_wait_all();
        __syncthreads();   // staged chunks visible to all waves

        v16h bk0 = *(const v16h*)(ldsK + (row)      * 32 + hi * 16);
        v16h bk1 = *(const v16h*)(ldsK + (16 + row) * 32 + hi * 16);
        v8f s0 = {}, s1 = {};
        s0 = __builtin_amdgcn_wmma_f32_16x16x32_f16(false, aq, false, bk0, (short)0, s0, false, false);
        s1 = __builtin_amdgcn_wmma_f32_16x16x32_f16(false, aq, false, bk1, (short)0, s1, false, false);

        // online softmax per row; C-layout: element r of a lane is row r+8*hi,
        // the 16 lanes of each half hold the 16 columns of that row.
#pragma unroll
        for (int r = 0; r < 8; ++r) {
            float v0 = s0[r] * scale, v1 = s1[r] * scale;
            float mx = fmaxf(v0, v1);
            mx = fmaxf(mx, __shfl_xor(mx, 1, 32));
            mx = fmaxf(mx, __shfl_xor(mx, 2, 32));
            mx = fmaxf(mx, __shfl_xor(mx, 4, 32));
            mx = fmaxf(mx, __shfl_xor(mx, 8, 32));
            float mnew = fmaxf(rmax[r], mx);
            float corr = __expf(rmax[r] - mnew);
            rmax[r] = mnew;
            float e0 = __expf(v0 - mnew), e1 = __expf(v1 - mnew);
            float ps = e0 + e1;
            ps += __shfl_xor(ps, 1, 32);
            ps += __shfl_xor(ps, 2, 32);
            ps += __shfl_xor(ps, 4, 32);
            ps += __shfl_xor(ps, 8, 32);
            rsum[r] = rsum[r] * corr + ps;
            o0[r] *= corr; o1[r] *= corr;
            s0[r] = e0; s1[r] = e1;
        }

        // C-layout -> A-layout relayout of P through wave-private LDS
#pragma unroll
        for (int r = 0; r < 8; ++r) {
            int rr = r + 8 * hi;
            pl[rr * 32 + row]      = (_Float16)s0[r];
            pl[rr * 32 + 16 + row] = (_Float16)s1[r];
        }
        __syncthreads();   // P visible (ds complete) across lanes
        v16h ap  = *(const v16h*)(pl + row * 32 + hi * 16);
        v16h bv0 = *(const v16h*)(ldsV + (row)      * 32 + hi * 16);
        v16h bv1 = *(const v16h*)(ldsV + (16 + row) * 32 + hi * 16);
        o0 = __builtin_amdgcn_wmma_f32_16x16x32_f16(false, ap, false, bv0, (short)0, o0, false, false);
        o1 = __builtin_amdgcn_wmma_f32_16x16x32_f16(false, ap, false, bv1, (short)0, o1, false, false);
        // WMMA consumes ds reads (s_wait_dscnt) before next loop-top barrier.
    }

    // finalize: /= rowsum, add positional encoding, store f16 GEMM rows
    const int b = ba >> 2, area = ba & 3;
#pragma unroll
    for (int r = 0; r < 8; ++r) {
        int n_local = n0 + r + 8 * hi;
        int n_global = area * 1024 + n_local;
        float inv = 1.0f / rsum[r];
        int c0 = h * 32 + row, c1 = c0 + 16;
        float pv0 = pp[((size_t)(b * 256 + c0) << 12) + n_global];
        float pv1 = pp[((size_t)(b * 256 + c1) << 12) + n_global];
        size_t base = ((size_t)(b * 4096 + n_global)) * 256;
        comb16[base + c0] = (_Float16)(o0[r] * inv + pv0);
        comb16[base + c1] = (_Float16)(o1[r] * inv + pv1);
    }
}

// ---------------------------------------------------------------------------
// Kernel F: projection GEMM: comb16 (8192x256) * wproj16^T + bias -> NCHW f32
// Weight slab async-staged in LDS as in kernel C.
// ---------------------------------------------------------------------------
__global__ __launch_bounds__(256) void proj_gemm_kernel(
    const _Float16* __restrict__ A, const _Float16* __restrict__ W,
    const float* __restrict__ bias, float* __restrict__ out)
{
    __shared__ _Float16 wlds[128 * 256];       // 64 KB weight slab
    const int lane = threadIdx.x & 31, wave = threadIdx.x >> 5;
    const int wm = wave & 3, wn = wave >> 2;
    const int m0 = blockIdx.x * 64 + wm * 16;
    const int n0 = wn * 64;
    const int nslab = blockIdx.y * 128;
    const int row = lane & 15, hi = lane >> 4;

    {
        const _Float16* wslab = W + (size_t)nslab * 256;
#pragma unroll
        for (int i = 0; i < 16; ++i) {
            int off = (threadIdx.x + i * 256) * 8;
            async_copy_b128(wslab + off, wlds + off);
        }
        async_wait_all();
        __syncthreads();
    }

    v8f acc[4] = {};
    const _Float16* arow = A + (size_t)(m0 + row) * 256 + hi * 16;
    for (int k0 = 0; k0 < 256; k0 += 32) {
        __builtin_prefetch(arow + k0 + 64, 0, 1);
        v16h a = *(const v16h*)(arow + k0);
#pragma unroll
        for (int t = 0; t < 4; ++t) {
            v16h bm = *(const v16h*)(wlds + (n0 + t * 16 + row) * 256 + k0 + hi * 16);
            acc[t] = __builtin_amdgcn_wmma_f32_16x16x32_f16(
                false, a, false, bm, (short)0, acc[t], false, false);
        }
    }
#pragma unroll
    for (int t = 0; t < 4; ++t) {
        int o = nslab + n0 + t * 16 + row;
        float bo = bias[o];
#pragma unroll
        for (int r = 0; r < 8; ++r) {
            int m = m0 + r + 8 * hi;
            int b = m >> 12, n = m & 4095;
            out[((size_t)(b * 256 + o) << 12) + n] = acc[t][r] + bo;
        }
    }
}

// ---------------------------------------------------------------------------
extern "C" void kernel_launch(void* const* d_in, const int* in_sizes, int n_in,
                              void* d_out, int out_size, void* d_ws, size_t ws_size,
                              hipStream_t stream)
{
    const float* x     = (const float*)d_in[0];
    const float* w_qk  = (const float*)d_in[1];
    const float* w_v   = (const float*)d_in[2];
    const float* w_pe  = (const float*)d_in[3];
    const float* w_pr  = (const float*)d_in[4];
    const float* qk_g  = (const float*)d_in[5];
    const float* qk_b  = (const float*)d_in[6];
    const float* qk_m  = (const float*)d_in[7];
    const float* qk_v  = (const float*)d_in[8];
    const float* v_g   = (const float*)d_in[9];
    const float* v_b   = (const float*)d_in[10];
    const float* v_m   = (const float*)d_in[11];
    const float* v_v   = (const float*)d_in[12];
    const float* pe_g  = (const float*)d_in[13];
    const float* pe_b  = (const float*)d_in[14];
    const float* pe_m  = (const float*)d_in[15];
    const float* pe_v  = (const float*)d_in[16];
    const float* pr_g  = (const float*)d_in[17];
    const float* pr_b  = (const float*)d_in[18];
    const float* pr_m  = (const float*)d_in[19];
    const float* pr_v  = (const float*)d_in[20];
    float* out = (float*)d_out;

    // workspace carve-up (256B aligned)
    char* ws = (char*)d_ws;
    size_t off = 0;
    auto carve = [&](size_t bytes) -> char* {
        char* p = ws + off;
        off = (off + bytes + 255) & ~(size_t)255;
        return p;
    };
    _Float16* wcat16  = (_Float16*)carve(768 * 256 * sizeof(_Float16));
    float*    biascat = (float*)   carve(768 * sizeof(float));
    _Float16* wproj16 = (_Float16*)carve(256 * 256 * sizeof(_Float16));
    float*    biaspr  = (float*)   carve(256 * sizeof(float));
    float*    wpe_s   = (float*)   carve(256 * 25 * sizeof(float));
    float*    biaspe  = (float*)   carve(256 * sizeof(float));
    _Float16* X16     = (_Float16*)carve((size_t)8192 * 256 * sizeof(_Float16));
    _Float16* q16     = (_Float16*)carve((size_t)8 * 8 * 1024 * 32 * sizeof(_Float16));
    _Float16* k16     = (_Float16*)carve((size_t)8 * 8 * 1024 * 32 * sizeof(_Float16));
    _Float16* v16t    = (_Float16*)carve((size_t)8 * 8 * 32 * 1024 * sizeof(_Float16));
    float*    vf32    = (float*)   carve((size_t)2 * 256 * 4096 * sizeof(float));
    float*    pp      = (float*)   carve((size_t)2 * 256 * 4096 * sizeof(float));
    _Float16* comb16  = (_Float16*)carve((size_t)8192 * 256 * sizeof(_Float16));

    fold_weights_kernel<<<256, 256, 0, stream>>>(
        w_qk, qk_g, qk_b, qk_m, qk_v,
        w_v,  v_g,  v_b,  v_m,  v_v,
        w_pe, pe_g, pe_b, pe_m, pe_v,
        w_pr, pr_g, pr_b, pr_m, pr_v,
        wcat16, biascat, wproj16, biaspr, wpe_s, biaspe);

    pack_x_kernel<<<(2 * 256 * 4096 + 255) / 256, 256, 0, stream>>>(x, X16);

    qkv_gemm_kernel<<<dim3(8192 / 64, 768 / 128), 256, 0, stream>>>(
        X16, wcat16, biascat, q16, k16, v16t, vf32);

    pe_conv_kernel<<<(2 * 256 * 4096 + 255) / 256, 256, 0, stream>>>(vf32, wpe_s, biaspe, pp);

    attn_kernel<<<dim3(8, 8, 8), 256, 0, stream>>>(q16, k16, v16t, pp, comb16);

    proj_gemm_kernel<<<dim3(8192 / 64, 256 / 128), 256, 0, stream>>>(
        comb16, wproj16, biaspr, out);
}
